// GlobalAttention_9715216023612
// MI455X (gfx1250) — matrix operand order repaired and checked by reading, compiled, not run on hardware
//
#include <hip/hip_runtime.h>
#include <hip/hip_bf16.h>
#include <math.h>

// ---------------------------------------------------------------------------
// GlobalAttention on MI455X (gfx1250): bf16 WMMA flash-attention with
// TDM (tensor_load_to_lds) double-buffered K/V staging.
// B=8, C=256, H=W=64, N=4096.
// ---------------------------------------------------------------------------

#define Bq  8
#define Cc  256
#define Nn  4096

typedef __attribute__((ext_vector_type(16))) __bf16 v16bf;
typedef __attribute__((ext_vector_type(8)))  __bf16 v8bf;
typedef __attribute__((ext_vector_type(8)))  float  v8f;
typedef __attribute__((ext_vector_type(4)))  unsigned int v4u;
typedef __attribute__((ext_vector_type(8)))  int v8i;
typedef __attribute__((ext_vector_type(4)))  int v4i;

union ABu { v16bf v; v8bf h[2]; };

// LDS row strides after TDM padding (elements)
#define KROW   272                 // 256 data + 16 pad (8 dwords)  = 544 B
#define VROW   72                  // 64 data + 8 pad (4 dwords)    = 144 B
#define KELEMS (64 * KROW)         // 17408 elems = 34816 B per buffer
#define VELEMS (Cc * VROW)         // 18432 elems = 36864 B per buffer
#define DYN_SMEM ((size_t)(2 * KELEMS + 2 * VELEMS) * 2)   // 143360 B

// ---------------------------------------------------------------------------
// TDM: 2D tile load Global -> LDS with row padding. D# per ISA ch.8.
// This toolchain's builtin takes 6 args: (g0, g1, g2, g3, g4, cpol).
// ---------------------------------------------------------------------------
__device__ __forceinline__ void tdm_load_2d(
    const void* lds_dst, const void* gsrc,
    unsigned tensor_d0, unsigned tensor_d1,
    unsigned tile_d0, unsigned tile_d1,
    unsigned stride0, unsigned pad_int, unsigned pad_amt)
{
    const unsigned long long ga = (unsigned long long)(uintptr_t)gsrc;
    v4u g0;
    g0.x = 1u;                                        // count=1, user D#
    g0.y = (unsigned)(uintptr_t)lds_dst;              // LDS byte address
    g0.z = (unsigned)ga;                              // global_addr[31:0]
    g0.w = ((unsigned)(ga >> 32) & 0x01FFFFFFu) | (2u << 30);  // [56:32]|type=2
    v8i g1;
    g1[0] = (int)((1u << 16) |                        // data_size = 2 bytes
                  (1u << 20) |                        // pad_enable
                  (pad_int << 22) | (pad_amt << 25));
    g1[1] = (int)((tensor_d0 & 0xFFFFu) << 16);                     // dim0 lo
    g1[2] = (int)(((tensor_d0 >> 16) & 0xFFFFu) |
                  ((tensor_d1 & 0xFFFFu) << 16));                   // dim0 hi | dim1 lo
    g1[3] = (int)(((tensor_d1 >> 16) & 0xFFFFu) |
                  ((tile_d0 & 0xFFFFu) << 16));                     // dim1 hi | tile0
    g1[4] = (int)(tile_d1 & 0xFFFFu);                               // tile1
    g1[5] = (int)stride0;                                           // stride0 lo
    g1[6] = 0;
    g1[7] = 0;
    v4i zz4 = {0, 0, 0, 0};
    v8i zz8 = {0, 0, 0, 0, 0, 0, 0, 0};
    __builtin_amdgcn_tensor_load_to_lds(g0, g1, zz4, zz4, zz8, 0);
}

// ---------------------------------------------------------------------------
// 1) Transpose x [B,C,N] f32 -> xt [B,N,C] bf16 (LDS 32x32 tiles)
// ---------------------------------------------------------------------------
__global__ __launch_bounds__(256) void transpose_x_kernel(
    const float* __restrict__ x, __bf16* __restrict__ xt)
{
    __shared__ float tile[32][33];
    const int c0 = blockIdx.x * 32, n0 = blockIdx.y * 32, b = blockIdx.z;
    const int tx = threadIdx.x, ty = threadIdx.y;          // (32,8)
    #pragma unroll
    for (int i = 0; i < 4; ++i) {
        const int cl = ty + i * 8;
        tile[cl][tx] = x[((long)b * Cc + c0 + cl) * Nn + n0 + tx];
    }
    __syncthreads();
    #pragma unroll
    for (int i = 0; i < 4; ++i) {
        const int nl = ty + i * 8;
        xt[((long)b * Nn + n0 + nl) * Cc + c0 + tx] = (__bf16)tile[tx][nl];
    }
}

// ---------------------------------------------------------------------------
// 2) Convert 3 weight matrices (256x256 f32) -> bf16, packed [3][256][256]
// ---------------------------------------------------------------------------
__global__ __launch_bounds__(256) void convert_w_kernel(
    const float* __restrict__ wq, const float* __restrict__ wk,
    const float* __restrict__ wv, __bf16* __restrict__ wb)
{
    const int i = blockIdx.x * 256 + threadIdx.x;
    const float* src = (blockIdx.y == 0) ? wq : (blockIdx.y == 1) ? wk : wv;
    wb[(long)blockIdx.y * (Cc * Cc) + i] = (__bf16)src[i];
}

// ---------------------------------------------------------------------------
// 3) Projection GEMM: q/k/v[b,n,d] = sum_c xt[b,n,c]*W[d,c] + bias[d]
//    q,k stored [B,N,C] bf16 ; v stored [B,C,N] bf16 (transposed scatter).
// ---------------------------------------------------------------------------
__global__ __launch_bounds__(256) void proj_kernel(
    const __bf16* __restrict__ xt, const __bf16* __restrict__ wb,
    const float* __restrict__ bqv, const float* __restrict__ bkv,
    const float* __restrict__ bvv,
    __bf16* __restrict__ qb, __bf16* __restrict__ kb, __bf16* __restrict__ vb)
{
    const int lane  = threadIdx.x & 31;
    const int wave  = threadIdx.x >> 5;
    const int row16 = lane & 15;
    const int hf    = lane >> 4;
    const long n0   = (long)blockIdx.x * 128 + wave * 16;   // global row in B*N

    ABu qa[8];
    const __bf16* arow = xt + (n0 + row16) * Cc;
    #pragma unroll
    for (int kk = 0; kk < 8; ++kk) {
        qa[kk].h[0] = *(const v8bf*)(arow + kk * 32 + 16 * hf);
        qa[kk].h[1] = *(const v8bf*)(arow + kk * 32 + 16 * hf + 8);
    }

    const int b  = (int)(n0 >> 12);
    const int nl = (int)(n0 & (Nn - 1));
    const float* biases[3] = { bqv, bkv, bvv };

    for (int o = 0; o < 3; ++o) {
        const __bf16* w = wb + (long)o * (Cc * Cc);
        for (int dt = 0; dt < 16; ++dt) {
            const int d = dt * 16 + row16;
            v8f acc = {};
            #pragma unroll
            for (int kk = 0; kk < 8; ++kk) {
                v16bf bf = *(const v16bf*)(w + (long)d * Cc + kk * 32 + 16 * hf);
                acc = __builtin_amdgcn_wmma_f32_16x16x32_bf16(
                    false, qa[kk].v, false, bf, (short)0, acc, false, false);
            }
            const float bia = biases[o][d];
            #pragma unroll
            for (int r = 0; r < 8; ++r) acc[r] += bia;

            if (o == 0 || o == 1) {          // q,k : [B,N,C]
                __bf16* dst = (o == 0) ? qb : kb;
                #pragma unroll
                for (int r = 0; r < 8; ++r)
                    dst[(n0 + r + 8 * hf) * Cc + d] = (__bf16)acc[r];
            } else {                          // v : [B,C,N]
                #pragma unroll
                for (int r = 0; r < 8; ++r)
                    vb[((long)b * Cc + d) * Nn + nl + r + 8 * hf] = (__bf16)acc[r];
            }
        }
    }
}

// ---------------------------------------------------------------------------
// 4) Per-(b,c) sum of squares over n for q and k  (ss[0..1][B][C])
// ---------------------------------------------------------------------------
__global__ __launch_bounds__(256) void sumsq_kernel(
    const __bf16* __restrict__ qb, const __bf16* __restrict__ kb,
    float* __restrict__ ss)
{
    const int b = blockIdx.x, which = blockIdx.y, c = threadIdx.x;
    const __bf16* src = ((which == 0) ? qb : kb) + (long)b * Nn * Cc;
    float acc = 0.f;
    for (int n = 0; n < Nn; ++n) {
        const float v = (float)src[(long)n * Cc + c];
        acc += v * v;
    }
    ss[((long)which * Bq + b) * Cc + c] = acc;
}

// ---------------------------------------------------------------------------
// 5) Fold L2 norms and 1/sqrt(N) into q,k:  q *= 1/(8*||q_c||), same for k.
// ---------------------------------------------------------------------------
__global__ __launch_bounds__(256) void scale_qk_kernel(
    __bf16* __restrict__ qb, __bf16* __restrict__ kb,
    const float* __restrict__ ss)
{
    const int n = blockIdx.x, b = blockIdx.y, c = threadIdx.x;
    const float nq = fmaxf(sqrtf(ss[(long)b * Cc + c]), 1e-12f);
    const float nk = fmaxf(sqrtf(ss[((long)Bq + b) * Cc + c]), 1e-12f);
    const long idx = ((long)b * Nn + n) * Cc + c;
    qb[idx] = (__bf16)((float)qb[idx] / (nq * 8.0f));
    kb[idx] = (__bf16)((float)kb[idx] / (nk * 8.0f));
}

// ---------------------------------------------------------------------------
// 6) Flash attention. Block = 128 thr = 4 waves; wave owns 16 query rows.
//    K/V tiles (64 m-cols) staged in LDS by TDM, double-buffered; wave 0
//    drives the DMA, one workgroup barrier per m-step. P^T via per-wave LDS.
// ---------------------------------------------------------------------------
__global__ __launch_bounds__(128) void attn_kernel(
    const __bf16* __restrict__ qb, const __bf16* __restrict__ kb,
    const __bf16* __restrict__ vb, const float* __restrict__ x,
    float* __restrict__ out)
{
    extern __shared__ __align__(128) char smem[];
    __bf16* lds_k = (__bf16*)smem;                         // 2 x KELEMS
    __bf16* lds_v = (__bf16*)(smem + 2 * KELEMS * 2);      // 2 x VELEMS
    __shared__ __attribute__((aligned(64))) __bf16 lds_p[4][16 * 64]; // [n][m]
    __shared__ __attribute__((aligned(64))) float  lds_f[4][16];

    const int lane = threadIdx.x & 31;
    const int wave = threadIdx.x >> 5;
    const int col  = lane & 15;
    const int hf   = lane >> 4;
    const int b    = blockIdx.y;
    const int n0   = blockIdx.x * 64 + wave * 16;

    const __bf16* qbase = qb + ((long)b * Nn + n0 + col) * Cc;  // A rows = n
    const __bf16* kbase = kb + (long)b * Nn * Cc;               // [m, c]
    const __bf16* vbase = vb + (long)b * Cc * Nn;               // [c, m]
    __bf16* pp = &lds_p[wave][0];
    float*  ff = &lds_f[wave][0];

    v8f o[16] = {};
    float mrun[8], lrun[8];
    #pragma unroll
    for (int r = 0; r < 8; ++r) { mrun[r] = -1e30f; lrun[r] = 0.f; }

    // Prologue: DMA first K/V tiles into buffer 0.
    if (wave == 0) {
        tdm_load_2d(lds_k, kbase, Cc, Nn, Cc, 64, Cc, 6, 7);
        tdm_load_2d(lds_v, vbase, Nn, Cc, 64, Cc, Nn, 4, 3);
    }

    for (int mb = 0; mb < Nn / 64; ++mb) {
        const int cur = mb & 1;
        if (wave == 0) __builtin_amdgcn_s_wait_tensorcnt(0);
        __syncthreads();   // tile[cur] ready; everyone done with tile[cur^1]
        if (wave == 0 && mb + 1 < Nn / 64) {
            const int m1 = (mb + 1) * 64;
            tdm_load_2d(lds_k + (cur ^ 1) * KELEMS, kbase + (long)m1 * Cc,
                        Cc, Nn, Cc, 64, Cc, 6, 7);
            tdm_load_2d(lds_v + (cur ^ 1) * VELEMS, vbase + m1,
                        Nn, Cc, 64, Cc, Nn, 4, 3);
        }
        const __bf16* kc = lds_k + cur * KELEMS;
        const __bf16* vc = lds_v + cur * VELEMS;

        // ---- S = q^T k : 16 rows x 64 cols, K=256 -> 32 wmma ----
        v8f s[4] = {};
        #pragma unroll
        for (int kk = 0; kk < 8; ++kk) {
            ABu a;
            a.h[0] = *(const v8bf*)(qbase + kk * 32 + 16 * hf);
            a.h[1] = *(const v8bf*)(qbase + kk * 32 + 16 * hf + 8);
            #pragma unroll
            for (int j = 0; j < 4; ++j) {
                v16bf kf = *(const v16bf*)(kc + (j * 16 + col) * KROW
                                              + kk * 32 + 16 * hf);
                s[j] = __builtin_amdgcn_wmma_f32_16x16x32_bf16(
                    false, a.v, false, kf, (short)0, s[j], false, false);
            }
        }

        // ---- online softmax over the 64 new columns ----
        #pragma unroll
        for (int r = 0; r < 8; ++r) {
            float t = fmaxf(fmaxf(s[0][r], s[1][r]), fmaxf(s[2][r], s[3][r]));
            t = fmaxf(t, __shfl_xor(t, 1, 16));
            t = fmaxf(t, __shfl_xor(t, 2, 16));
            t = fmaxf(t, __shfl_xor(t, 4, 16));
            t = fmaxf(t, __shfl_xor(t, 8, 16));
            const float mnew = fmaxf(mrun[r], t);
            const float fr   = __expf(mrun[r] - mnew);
            mrun[r] = mnew;
            float rs = 0.f;
            #pragma unroll
            for (int j = 0; j < 4; ++j) {
                const float p = __expf(s[j][r] - mnew);
                rs += p;
                pp[(r + 8 * hf) * 64 + j * 16 + col] = (__bf16)p;  // P^T stage
            }
            rs += __shfl_xor(rs, 1, 16);
            rs += __shfl_xor(rs, 2, 16);
            rs += __shfl_xor(rs, 4, 16);
            rs += __shfl_xor(rs, 8, 16);
            lrun[r] = lrun[r] * fr + rs;
            if (col == 0) ff[8 * hf + r] = fr;   // lanes 0,16 publish factors
        }
        asm volatile("s_wait_dscnt 0x0" ::: "memory");
        const float fo = ff[col];

        // ---- Ot = Ot*fo + V * P^T : 16 c-tiles x K=64 -> 32 wmma ----
        #pragma unroll
        for (int ct = 0; ct < 16; ++ct) {
            v8f acc = o[ct];
            #pragma unroll
            for (int r = 0; r < 8; ++r) acc[r] *= fo;
            #pragma unroll
            for (int kj = 0; kj < 2; ++kj) {
                ABu va;                                   // A rows = c, K = m
                const __bf16* vrow = vc + (ct * 16 + col) * VROW + kj * 32;
                va.h[0] = *(const v8bf*)(vrow + 16 * hf);
                va.h[1] = *(const v8bf*)(vrow + 16 * hf + 8);
                v16bf pf = *(const v16bf*)(pp + col * 64 + kj * 32 + 16 * hf);
                acc = __builtin_amdgcn_wmma_f32_16x16x32_bf16(
                    false, va.v, false, pf, (short)0, acc, false, false);
            }
            o[ct] = acc;
        }
    }

    // ---- finalize: divide by row sums, add residual, store [B,C,N] ----
    if (col == 0) {
        #pragma unroll
        for (int r = 0; r < 8; ++r) ff[8 * hf + r] = lrun[r];
    }
    asm volatile("s_wait_dscnt 0x0" ::: "memory");
    const float linv = 1.0f / ff[col];
    const long xoff = (long)b * Cc * Nn;
    #pragma unroll
    for (int ct = 0; ct < 16; ++ct) {
        #pragma unroll
        for (int r = 0; r < 8; ++r) {
            const long c   = ct * 16 + r + 8 * hf;
            const long idx = xoff + c * Nn + n0 + col;
            out[idx] = x[idx] + o[ct][r] * linv;
        }
    }
}

// ---------------------------------------------------------------------------
// Launch
// ---------------------------------------------------------------------------
extern "C" void kernel_launch(void* const* d_in, const int* in_sizes, int n_in,
                              void* d_out, int out_size, void* d_ws, size_t ws_size,
                              hipStream_t stream)
{
    (void)in_sizes; (void)n_in; (void)out_size; (void)ws_size;
    const float* x   = (const float*)d_in[0];
    const float* wq  = (const float*)d_in[1];
    const float* bqp = (const float*)d_in[2];
    const float* wk  = (const float*)d_in[3];
    const float* bkp = (const float*)d_in[4];
    const float* wv  = (const float*)d_in[5];
    const float* bvp = (const float*)d_in[6];
    float* out = (float*)d_out;

    char* ws = (char*)d_ws;
    size_t off = 0;
    auto carve = [&](size_t bytes) -> void* {
        void* p = ws + off;
        off += (bytes + 255) & ~(size_t)255;
        return p;
    };
    const size_t tens = (size_t)Bq * Nn * Cc * sizeof(__bf16);  // 16.78 MB
    __bf16* xt = (__bf16*)carve(tens);
    __bf16* qb = (__bf16*)carve(tens);
    __bf16* kb = (__bf16*)carve(tens);
    __bf16* vb = (__bf16*)carve(tens);
    __bf16* wb = (__bf16*)carve((size_t)3 * Cc * Cc * sizeof(__bf16));
    float*  ss = (float*)carve((size_t)2 * Bq * Cc * sizeof(float));

    transpose_x_kernel<<<dim3(Cc / 32, Nn / 32, Bq), dim3(32, 8), 0, stream>>>(x, xt);
    convert_w_kernel<<<dim3(Cc * Cc / 256, 3), 256, 0, stream>>>(wq, wk, wv, wb);
    proj_kernel<<<dim3(Bq * Nn / 128), 256, 0, stream>>>(xt, wb, bqp, bkp, bvp,
                                                         qb, kb, vb);
    sumsq_kernel<<<dim3(Bq, 2), 256, 0, stream>>>(qb, kb, ss);
    scale_qk_kernel<<<dim3(Nn, Bq), 256, 0, stream>>>(qb, kb, ss);
    attn_kernel<<<dim3(Nn / 64, Bq), 128, DYN_SMEM, stream>>>(qb, kb, vb, x, out);
}
